// WindowAttention_79688823210190
// MI455X (gfx1250) — compile-verified, matrix operand order checked by reading, not run on hardware
//
#include <hip/hip_runtime.h>

// ---------------------------------------------------------------------------
// CDNA5 (gfx1250) window-attention: bf16 WMMA (16x16x32) with f32 accumulate.
// Double-buffered LDS GEMM pipelines; GLOBAL_LOAD_ASYNC_TO_LDS_B128 for the
// pure-bf16 copies (proj A-tiles, attention Q/K tiles), ASYNCcnt fenced.
// fp32->bf16 staging uses v_cvt_pk_bf16_f32 (packed, 1 VALU op per 2 elems).
// ws layout (bf16 elements):
//   [0)              q   : 1024*16*64*32 = 33554432 elems (64 MiB)
//   [33554432)       k   : 33554432
//   [67108864)       v   : 33554432
//   [100663296)      ao  : attention output, [m=b*64+tok][c=h*32+dd], 33554432
// total 256 MiB of d_ws.
// ---------------------------------------------------------------------------

typedef __attribute__((ext_vector_type(8)))  unsigned short v8u;
typedef __attribute__((ext_vector_type(16))) unsigned short v16u;
typedef __attribute__((ext_vector_type(16))) __bf16        v16bf;
typedef __attribute__((ext_vector_type(8)))  float         v8f;

#define QKV_ELEMS 33554432ull  // 1024*16*64*32

static __device__ __forceinline__ unsigned short f2bf(float f) {
  // round-to-nearest-even float -> bf16 (scalar fallback / epilogue stores)
  unsigned int u = __builtin_bit_cast(unsigned int, f);
  u += 0x7fffu + ((u >> 16) & 1u);
  return (unsigned short)(u >> 16);
}

// Two floats -> packed bf16x2 in one dword (v_cvt_pk_bf16_f32 when available).
static __device__ __forceinline__ unsigned int pack_bf16x2(float lo, float hi) {
#if __has_builtin(__builtin_amdgcn_cvt_pk_bf16_f32)
  auto r = __builtin_amdgcn_cvt_pk_bf16_f32(lo, hi);
  return __builtin_bit_cast(unsigned int, r);
#else
  return (unsigned int)f2bf(lo) | ((unsigned int)f2bf(hi) << 16);
#endif
}

static __device__ __forceinline__ v8f zero_v8f() {
  v8f z = {0.f, 0.f, 0.f, 0.f, 0.f, 0.f, 0.f, 0.f};
  return z;
}

// Low 32 bits of a generic pointer to __shared__ == LDS byte address
// (flat LDS aperture maps addr[31:0] to LDS space per the ISA).
static __device__ __forceinline__ unsigned int lds_addr32(const void* p) {
  return (unsigned int)(unsigned long long)p;
}

// Async 16-byte global->LDS copy (GVS form: SGPR64 base + per-lane u32 offset).
// Bytes bypass VGPRs; completion tracked by ASYNCcnt.
static __device__ __forceinline__ void async_cp_b128(unsigned int lds_off,
                                                     unsigned int goff,
                                                     unsigned long long sbase) {
  asm volatile("global_load_async_to_lds_b128 %0, %1, %2"
               :
               : "v"(lds_off), "v"(goff), "s"(sbase)
               : "memory");
}

static __device__ __forceinline__ void wait_asynccnt0() {
  asm volatile("s_wait_asynccnt 0x0" ::: "memory");
}

// Load one 16-bit A/B fragment for this lane from LDS.
// p points at row_base + half*8 halves; the two 8-half chunks sit at
// K offsets {half*8, 16+half*8} per the CDNA5 16-bit fragment layout.
static __device__ __forceinline__ v16bf ld_frag(const unsigned short* p) {
  v8u lo = *(const v8u*)(p);
  v8u hi = *(const v8u*)(p + 16);
  v16u c = __builtin_shufflevector(lo, hi, 0, 1, 2, 3, 4, 5, 6, 7,
                                   8, 9, 10, 11, 12, 13, 14, 15);
  return __builtin_bit_cast(v16bf, c);
}

static __device__ __forceinline__ v8f wmma_bf16(v16bf a, v16bf b, v8f c) {
  return __builtin_amdgcn_wmma_f32_16x16x32_bf16(false, a, false, b, (short)0,
                                                 c, false, false);
}

// ---------------------------------------------------------------------------
// Kernel 1: QKV projection. out[m][n3] = x[m][:] . w_qkv[n3][:] + b_qkv[n3]
// M=65536, N=1536, K=512. Block tile 128x128x32, 8 waves (2x4), each wave
// computes 64x32 (4x2 WMMA tiles). Double-buffered LDS (one barrier/K-step).
// fp32 -> bf16 conversion happens while staging (packed cvt), VGPR path.
// Epilogue scatters to [which][b][h][tok][dd] bf16 in ws.
// ---------------------------------------------------------------------------
__global__ __launch_bounds__(256, 2) void qkv_gemm_kernel(
    const float* __restrict__ x, const float* __restrict__ wqkv,
    const float* __restrict__ bqkv, unsigned short* __restrict__ qkv) {
  __shared__ unsigned short As[2][128 * 40];  // [buf][row m][k], stride 40
  __shared__ unsigned short Bs[2][128 * 40];  // [buf][col n][k], stride 40
  const int tid = threadIdx.x;
  const int lane = tid & 31;
  const int wave = tid >> 5;
  const int wm = wave >> 2;  // 0..1
  const int wn = wave & 3;   // 0..3
  const int hl = lane >> 4;
  const int l16 = lane & 15;
  const int bm = blockIdx.y * 128;
  const int bn = blockIdx.x * 128;

  v8f acc[4][2];
#pragma unroll
  for (int mi = 0; mi < 4; ++mi)
#pragma unroll
    for (int ni = 0; ni < 2; ++ni) acc[mi][ni] = zero_v8f();

  const int cg = tid & 7;   // 8 groups of 4 floats per 32-wide row
  const int r0 = tid >> 3;  // 32 rows per pass

  auto stage = [&](int buf, int k0) {
#pragma unroll
    for (int it = 0; it < 4; ++it) {
      int r = r0 + 32 * it;
      float4 a = *(const float4*)&x[(size_t)(bm + r) * 512 + k0 + cg * 4];
      uint2 ua = {pack_bf16x2(a.x, a.y), pack_bf16x2(a.z, a.w)};
      *(uint2*)&As[buf][r * 40 + cg * 4] = ua;
      float4 b = *(const float4*)&wqkv[(size_t)(bn + r) * 512 + k0 + cg * 4];
      uint2 ub = {pack_bf16x2(b.x, b.y), pack_bf16x2(b.z, b.w)};
      *(uint2*)&Bs[buf][r * 40 + cg * 4] = ub;
    }
  };

  stage(0, 0);
  for (int kt = 0; kt < 16; ++kt) {
    const int cur = kt & 1;
    __syncthreads();  // stage(kt) visible; prior reads of buf cur^1 done
    v16bf fa[4], fb[2];
#pragma unroll
    for (int mi = 0; mi < 4; ++mi)
      fa[mi] = ld_frag(&As[cur][(wm * 64 + mi * 16 + l16) * 40 + hl * 8]);
#pragma unroll
    for (int ni = 0; ni < 2; ++ni)
      fb[ni] = ld_frag(&Bs[cur][(wn * 32 + ni * 16 + l16) * 40 + hl * 8]);
    if (kt + 1 < 16) stage(cur ^ 1, (kt + 1) * 32);  // overlap with WMMA
#pragma unroll
    for (int mi = 0; mi < 4; ++mi)
#pragma unroll
      for (int ni = 0; ni < 2; ++ni)
        acc[mi][ni] = wmma_bf16(fa[mi], fb[ni], acc[mi][ni]);
  }

#pragma unroll
  for (int mi = 0; mi < 4; ++mi) {
#pragma unroll
    for (int ni = 0; ni < 2; ++ni) {
      int gn = bn + wn * 32 + ni * 16 + l16;  // 0..1535
      float bias = bqkv[gn];
      int which = gn >> 9;      // 0=q, 1=k, 2=v
      int hh = (gn >> 5) & 15;  // head
      int dd = gn & 31;         // dim within head
#pragma unroll
      for (int vv = 0; vv < 8; ++vv) {
        int gm = bm + wm * 64 + mi * 16 + vv + 8 * hl;
        int b = gm >> 6;
        int tok = gm & 63;
        size_t dst = (size_t)which * QKV_ELEMS +
                     ((size_t)(b * 16 + hh) * 64 + tok) * 32 + dd;
        qkv[dst] = f2bf(acc[mi][ni][vv] + bias);
      }
    }
  }
}

// ---------------------------------------------------------------------------
// Kernel 2: attention per (window b, head h). Block = 4 waves; wave w owns
// query rows [16w, 16w+16). Q/K tiles staged with async global->LDS b128
// copies; V goes through VGPRs (needs an in-flight transpose). S = (Q Kt)*
// scale + bias, softmax rows via 16-lane shfl_xor reductions, P->LDS bf16,
// O = P V via WMMA.
// ---------------------------------------------------------------------------
__global__ __launch_bounds__(128, 4) void attn_kernel(
    const unsigned short* __restrict__ qkv, const int* __restrict__ relidx,
    const float* __restrict__ table, unsigned short* __restrict__ ao) {
  __shared__ unsigned short Qs[64 * 40];  // [tok][d]
  __shared__ unsigned short Ks[64 * 40];  // [tok][d]
  __shared__ unsigned short Vt[32 * 72];  // [d][tok] (transposed)
  __shared__ unsigned short Ps[64 * 72];  // [i][j]
  const int tid = threadIdx.x;
  const int lane = tid & 31;
  const int wave = tid >> 5;
  const int hl = lane >> 4;
  const int l16 = lane & 15;
  const int bh = blockIdx.x;  // b*16 + h
  const int b = bh >> 4;
  const int h = bh & 15;
  const int widx = b & 63;                // window index = b % nW
  const size_t base = (size_t)bh * 2048;  // 64 tok * 32 d
  const unsigned short* qg = qkv + base;
  const unsigned short* kg = qkv + QKV_ELEMS + base;
  const unsigned short* vg = qkv + 2 * QKV_ELEMS + base;
  const unsigned long long qb = (unsigned long long)qg;
  const unsigned long long kb = (unsigned long long)kg;

#pragma unroll
  for (int it = 0; it < 2; ++it) {
    int idx = tid + it * 128;  // 0..255 -> 64 rows x 4 groups of 8 halves
    int row = idx >> 2;
    int cg = idx & 3;
    unsigned int goff = (unsigned int)(row * 64 + cg * 16);  // bytes
    async_cp_b128(lds_addr32(&Qs[row * 40 + cg * 8]), goff, qb);
    async_cp_b128(lds_addr32(&Ks[row * 40 + cg * 8]), goff, kb);
    uint4 vvv = *(const uint4*)&vg[row * 32 + cg * 8];
    Vt[(cg * 8 + 0) * 72 + row] = (unsigned short)(vvv.x & 0xffffu);
    Vt[(cg * 8 + 1) * 72 + row] = (unsigned short)(vvv.x >> 16);
    Vt[(cg * 8 + 2) * 72 + row] = (unsigned short)(vvv.y & 0xffffu);
    Vt[(cg * 8 + 3) * 72 + row] = (unsigned short)(vvv.y >> 16);
    Vt[(cg * 8 + 4) * 72 + row] = (unsigned short)(vvv.z & 0xffffu);
    Vt[(cg * 8 + 5) * 72 + row] = (unsigned short)(vvv.z >> 16);
    Vt[(cg * 8 + 6) * 72 + row] = (unsigned short)(vvv.w & 0xffffu);
    Vt[(cg * 8 + 7) * 72 + row] = (unsigned short)(vvv.w >> 16);
  }
  wait_asynccnt0();
  __syncthreads();

  // S = Q . K^T  (K = d = 32, one WMMA per 16-wide j tile)
  v16bf aq = ld_frag(&Qs[(wave * 16 + l16) * 40 + hl * 8]);
  v8f s[4];
#pragma unroll
  for (int t = 0; t < 4; ++t) {
    v16bf bk = ld_frag(&Ks[(t * 16 + l16) * 40 + hl * 8]);
    s[t] = wmma_bf16(aq, bk, zero_v8f());
  }

  const float scale = 0.17677669529663689f;  // 32^-0.5
  const int* ri = relidx + widx * 4096;
  float p[4][8];
#pragma unroll
  for (int vv = 0; vv < 8; ++vv) {
    int i = wave * 16 + vv + 8 * hl;  // query row
    float mx = -3.4e38f;
#pragma unroll
    for (int t = 0; t < 4; ++t) {
      int j = t * 16 + l16;
      int idx = ri[i * 64 + j];
      float sv = s[t][vv] * scale + table[idx * 16 + h];
      p[t][vv] = sv;
      mx = fmaxf(mx, sv);
    }
    mx = fmaxf(mx, __shfl_xor(mx, 1, 32));
    mx = fmaxf(mx, __shfl_xor(mx, 2, 32));
    mx = fmaxf(mx, __shfl_xor(mx, 4, 32));
    mx = fmaxf(mx, __shfl_xor(mx, 8, 32));
    float sum = 0.f;
#pragma unroll
    for (int t = 0; t < 4; ++t) {
      float e = __expf(p[t][vv] - mx);
      p[t][vv] = e;
      sum += e;
    }
    sum += __shfl_xor(sum, 1, 32);
    sum += __shfl_xor(sum, 2, 32);
    sum += __shfl_xor(sum, 4, 32);
    sum += __shfl_xor(sum, 8, 32);
    float inv = 1.0f / sum;
#pragma unroll
    for (int t = 0; t < 4; ++t) {
      int j = t * 16 + l16;
      Ps[i * 72 + j] = f2bf(p[t][vv] * inv);
    }
  }
  // wave reads back only its own Ps rows: in-wave DS ordering, no barrier.

  // O = P . V   (M=16 rows of this wave, N=32, K=64 -> 2 K steps x 2 n tiles)
  v8f o[2] = {zero_v8f(), zero_v8f()};
#pragma unroll
  for (int kk = 0; kk < 2; ++kk) {
    v16bf ap = ld_frag(&Ps[(wave * 16 + l16) * 72 + kk * 32 + hl * 8]);
#pragma unroll
    for (int dt = 0; dt < 2; ++dt) {
      v16bf bv = ld_frag(&Vt[(dt * 16 + l16) * 72 + kk * 32 + hl * 8]);
      o[dt] = wmma_bf16(ap, bv, o[dt]);
    }
  }
#pragma unroll
  for (int dt = 0; dt < 2; ++dt)
#pragma unroll
    for (int vv = 0; vv < 8; ++vv) {
      int i = wave * 16 + vv + 8 * hl;
      int c = h * 32 + dt * 16 + l16;
      ao[(size_t)(b * 64 + i) * 512 + c] = f2bf(o[dt][vv]);
    }
}

// ---------------------------------------------------------------------------
// Kernel 3: output projection. out[m][n] = ao[m][:] . w_proj[n][:] + b_proj[n]
// M=65536, N=512, K=512. Double-buffered; A tile is already bf16 in ws, so it
// is staged with async global->LDS b128 copies (no VGPR round-trip); B tile
// converts fp32 w_proj -> bf16 (packed cvt) through VGPRs.
// ---------------------------------------------------------------------------
__global__ __launch_bounds__(256, 2) void proj_gemm_kernel(
    const unsigned short* __restrict__ ao, const float* __restrict__ wproj,
    const float* __restrict__ bproj, float* __restrict__ out) {
  __shared__ unsigned short As[2][128 * 40];
  __shared__ unsigned short Bs[2][128 * 40];
  const int tid = threadIdx.x;
  const int lane = tid & 31;
  const int wave = tid >> 5;
  const int wm = wave >> 2;
  const int wn = wave & 3;
  const int hl = lane >> 4;
  const int l16 = lane & 15;
  const int bm = blockIdx.y * 128;
  const int bn = blockIdx.x * 128;
  const unsigned long long ab = (unsigned long long)ao;

  v8f acc[4][2];
#pragma unroll
  for (int mi = 0; mi < 4; ++mi)
#pragma unroll
    for (int ni = 0; ni < 2; ++ni) acc[mi][ni] = zero_v8f();

  const int cg8 = tid & 3;   // A: 4 groups of 8 bf16 per 32-wide row
  const int ra0 = tid >> 2;  // 64 rows per pass
  const int cg4 = tid & 7;   // B: 8 groups of 4 floats
  const int rb0 = tid >> 3;  // 32 rows per pass

  auto stageA = [&](int buf, int k0) {
#pragma unroll
    for (int it = 0; it < 2; ++it) {
      int r = ra0 + 64 * it;
      unsigned int goff = (unsigned int)((bm + r) * 1024 + k0 * 2 + cg8 * 16);
      async_cp_b128(lds_addr32(&As[buf][r * 40 + cg8 * 8]), goff, ab);
    }
  };
  auto stageB = [&](int buf, int k0) {
#pragma unroll
    for (int it = 0; it < 4; ++it) {
      int r = rb0 + 32 * it;
      float4 bw = *(const float4*)&wproj[(size_t)(bn + r) * 512 + k0 + cg4 * 4];
      uint2 ub = {pack_bf16x2(bw.x, bw.y), pack_bf16x2(bw.z, bw.w)};
      *(uint2*)&Bs[buf][r * 40 + cg4 * 4] = ub;
    }
  };

  stageA(0, 0);
  stageB(0, 0);
  for (int kt = 0; kt < 16; ++kt) {
    const int cur = kt & 1;
    wait_asynccnt0();  // async stage of tile kt complete (after prev WMMAs)
    __syncthreads();
    v16bf fa[4], fb[2];
#pragma unroll
    for (int mi = 0; mi < 4; ++mi)
      fa[mi] = ld_frag(&As[cur][(wm * 64 + mi * 16 + l16) * 40 + hl * 8]);
#pragma unroll
    for (int ni = 0; ni < 2; ++ni)
      fb[ni] = ld_frag(&Bs[cur][(wn * 32 + ni * 16 + l16) * 40 + hl * 8]);
    if (kt + 1 < 16) {
      stageA(cur ^ 1, (kt + 1) * 32);
      stageB(cur ^ 1, (kt + 1) * 32);
    }
#pragma unroll
    for (int mi = 0; mi < 4; ++mi)
#pragma unroll
      for (int ni = 0; ni < 2; ++ni)
        acc[mi][ni] = wmma_bf16(fa[mi], fb[ni], acc[mi][ni]);
  }

#pragma unroll
  for (int mi = 0; mi < 4; ++mi) {
#pragma unroll
    for (int ni = 0; ni < 2; ++ni) {
      int gn = bn + wn * 32 + ni * 16 + l16;
      float bias = bproj[gn];
#pragma unroll
      for (int vv = 0; vv < 8; ++vv) {
        int gm = bm + wm * 64 + mi * 16 + vv + 8 * hl;
        out[(size_t)gm * 512 + gn] = acc[mi][ni][vv] + bias;
      }
    }
  }
}

// ---------------------------------------------------------------------------
extern "C" void kernel_launch(void* const* d_in, const int* in_sizes, int n_in,
                              void* d_out, int out_size, void* d_ws,
                              size_t ws_size, hipStream_t stream) {
  (void)in_sizes; (void)n_in; (void)out_size; (void)ws_size;
  const float* x      = (const float*)d_in[0];
  const int*   relidx = (const int*)d_in[1];
  const float* wqkv   = (const float*)d_in[2];
  const float* bqkv   = (const float*)d_in[3];
  const float* table  = (const float*)d_in[4];
  const float* wproj  = (const float*)d_in[5];
  const float* bproj  = (const float*)d_in[6];
  unsigned short* ws  = (unsigned short*)d_ws;
  unsigned short* qkv = ws;                     // q,k,v (3 * QKV_ELEMS bf16)
  unsigned short* ao  = ws + 3ull * QKV_ELEMS;  // attention output bf16
  float* out = (float*)d_out;

  // QKV GEMM: M tiles = 65536/128 = 512, N tiles = 1536/128 = 12
  qkv_gemm_kernel<<<dim3(12, 512), 256, 0, stream>>>(x, wqkv, bqkv, qkv);
  // one block per (window, head)
  attn_kernel<<<dim3(1024 * 16), 128, 0, stream>>>(qkv, relidx, table, ao);
  // Proj GEMM: N tiles = 512/128 = 4
  proj_gemm_kernel<<<dim3(4, 512), 256, 0, stream>>>(ao, wproj, bproj, out);
}
